// HMM_80470507258177
// MI455X (gfx1250) — matrix operand order repaired and checked by reading, not compiled, not run on hardware
//
#include <hip/hip_runtime.h>
#include <hip/hip_bf16.h>

typedef __attribute__((ext_vector_type(16))) _Float16 v16h;
typedef __attribute__((ext_vector_type(8)))  _Float16 v8h;
typedef __attribute__((ext_vector_type(8)))  float    v8f;

#define HMM_B    128
#define HMM_TMAX 2048
#define HMM_S    512
#define HMM_O    1024
#define USTRIDE  520   // halves: 512 + 8 pad -> conflict-free 16B lane loads (stride mult. of 8)
#define ASTRIDE  516   // f32:    512 + 4 pad -> half-waves hit disjoint bank groups on writeback

__device__ __forceinline__ float wave_max(float v) {
#pragma unroll
    for (int off = 16; off > 0; off >>= 1) v = fmaxf(v, __shfl_xor(v, off, 32));
    return v;
}
__device__ __forceinline__ float wave_sum(float v) {
#pragma unroll
    for (int off = 16; off > 0; off >>= 1) v += __shfl_xor(v, off, 32);
    return v;
}

// ---------- precompute: log_softmax(pi) ----------
__global__ __launch_bounds__(512)
void prep_logpi(const float* __restrict__ pi, float* __restrict__ logpi) {
    __shared__ float part[16];
    const int tid = threadIdx.x, lane = tid & 31, w = tid >> 5;
    const float v = pi[tid];
    float mx = wave_max(v);
    if (lane == 0) part[w] = mx;
    __syncthreads();
    float m = part[0];
#pragma unroll
    for (int i = 1; i < 16; ++i) m = fmaxf(m, part[i]);
    __syncthreads();
    float s = wave_sum(__expf(v - m));
    if (lane == 0) part[w] = s;
    __syncthreads();
    float tot = 0.f;
#pragma unroll
    for (int i = 0; i < 16; ++i) tot += part[i];
    logpi[tid] = v - m - __logf(tot);
}

// ---------- precompute: A = softmax(transition, axis=0) as f16, row-major [s_next][s_prev] ----------
__global__ __launch_bounds__(512)
void prep_trans(const float* __restrict__ trans, _Float16* __restrict__ Af16) {
    const int lane = threadIdx.x & 31, w = threadIdx.x >> 5;
    const int col = blockIdx.x * 16 + w;          // s_prev column; reduce over s_next rows
    float v[16];
    float mx = -3.0e38f;
#pragma unroll
    for (int j = 0; j < 16; ++j) {
        v[j] = trans[(size_t)(lane + 32 * j) * HMM_S + col];
        mx = fmaxf(mx, v[j]);
    }
    mx = wave_max(mx);
    float se = 0.f;
#pragma unroll
    for (int j = 0; j < 16; ++j) se += __expf(v[j] - mx);
    se = wave_sum(se);
    const float inv = 1.0f / se;
#pragma unroll
    for (int j = 0; j < 16; ++j)
        Af16[(size_t)(lane + 32 * j) * HMM_S + col] = (_Float16)(__expf(v[j] - mx) * inv);
}

// ---------- precompute: logEt[o][s] = log_softmax(emission, axis=1) TRANSPOSED ----------
__global__ __launch_bounds__(256)
void prep_emis(const float* __restrict__ em, float* __restrict__ logEt) {
    __shared__ float part[8];
    const int tid = threadIdx.x, lane = tid & 31, w = tid >> 5;
    const int s = blockIdx.x;                      // state row of emission
    const float* row = em + (size_t)s * HMM_O;
    float v[4];
    float mx = -3.0e38f;
#pragma unroll
    for (int j = 0; j < 4; ++j) { v[j] = row[tid + 256 * j]; mx = fmaxf(mx, v[j]); }
    mx = wave_max(mx);
    if (lane == 0) part[w] = mx;
    __syncthreads();
    float m = part[0];
#pragma unroll
    for (int i = 1; i < 8; ++i) m = fmaxf(m, part[i]);
    __syncthreads();
    float se = 0.f;
#pragma unroll
    for (int j = 0; j < 4; ++j) se += __expf(v[j] - m);
    se = wave_sum(se);
    if (lane == 0) part[w] = se;
    __syncthreads();
    float tot = 0.f;
#pragma unroll
    for (int i = 0; i < 8; ++i) tot += part[i];
    const float lz = m + __logf(tot);
#pragma unroll
    for (int j = 0; j < 4; ++j)
        logEt[(size_t)(tid + 256 * j) * HMM_S + s] = v[j] - lz;  // transposed write (one-time)
}

// ---------- main: HMM forward, 16 batch rows per workgroup, WMMA per step ----------
__global__ __launch_bounds__(512)
void hmm_forward(const int* __restrict__ x, const int* __restrict__ Tlen,
                 const float* __restrict__ logpi, const _Float16* __restrict__ Af16,
                 const float* __restrict__ logEt, float* __restrict__ out) {
    __shared__ float    alpha_sh[16 * ASTRIDE];    // ~33 KB
    __shared__ _Float16 U_sh[16 * USTRIDE];        // ~16.3 KB
    __shared__ float    m_sh[16];
    __shared__ int      xt_sh[16];

    const int tid   = threadIdx.x;
    const int lane  = tid & 31;
    const int w     = tid >> 5;                    // 16 waves; wave w owns batch row w
    const int bbase = blockIdx.x * 16;

    // workgroup-uniform max sequence length over its 16 rows
    int tmax = 1;
#pragma unroll
    for (int i = 0; i < 16; ++i) {
        const int tv = Tlen[bbase + i];
        tmax = (tv > tmax) ? tv : tmax;
    }
    const int myT = Tlen[bbase + w];

    // alpha_0[b,s] = logEt[x[b,0], s] + logpi[s]   (contiguous in s)
    {
        const int x0 = x[(size_t)(bbase + w) * HMM_TMAX];
        const float* er = logEt + (size_t)x0 * HMM_S;
#pragma unroll
        for (int j = 0; j < 16; ++j) {
            const int c = lane + 32 * j;
            alpha_sh[w * ASTRIDE + c] = er[c] + logpi[c];
        }
    }
    __syncthreads();

    for (int t = 0;; ++t) {
        // ---- phase 1: per-row max / sumexp; emit output at t == T[b]-1; U = exp(alpha-m) in f16
        float v[16];
        float mx = -3.0e38f;
#pragma unroll
        for (int j = 0; j < 16; ++j) {
            v[j] = alpha_sh[w * ASTRIDE + lane + 32 * j];
            mx = fmaxf(mx, v[j]);
        }
        mx = wave_max(mx);
        float se = 0.f;
#pragma unroll
        for (int j = 0; j < 16; ++j) se += __expf(v[j] - mx);
        se = wave_sum(se);
        if (lane == 0) {
            m_sh[w] = mx;
            if (myT - 1 == t) out[bbase + w] = mx + __logf(se);
            if (t + 1 < tmax) xt_sh[w] = x[(size_t)(bbase + w) * HMM_TMAX + t + 1];
        }
#pragma unroll
        for (int j = 0; j < 16; ++j)
            U_sh[w * USTRIDE + lane + 32 * j] = (_Float16)__expf(v[j] - mx);
        __syncthreads();

        if (t == tmax - 1) break;   // uniform across workgroup

        // ---- phase 2: V = U @ A^T via v_wmma_f32_16x16x32_f16
        // wave w covers state cols [32w, 32w+32): two fused N-tiles sharing the A-fragment,
        // each with even/odd K accumulators -> 4 independent WMMA chains of length 8.
        const int arow  = lane & 15;
        const int koff  = (lane >> 4) * 16;        // A-frag: lanes>=16 hold K offsets +16
        const int ncol0 = w * 32 + (lane & 15);
        const int ncol1 = ncol0 + 16;
        const _Float16* bp0 = Af16 + (size_t)ncol0 * HMM_S + (lane >> 4) * 16; // B-frag: lanes>=16 K+16
        const _Float16* bp1 = Af16 + (size_t)ncol1 * HMM_S + (lane >> 4) * 16;
        const _Float16* up  = &U_sh[arow * USTRIDE + koff];

        v8f acc0a = {}, acc0b = {}, acc1a = {}, acc1b = {};
#pragma unroll 2
        for (int kt = 0; kt < 16; kt += 2) {
            {   // even K-step -> chains a
                const v8h lo = *(const v8h*)(up + kt * 32);
                const v8h hi = *(const v8h*)(up + kt * 32 + 8);
                v16h a;
#pragma unroll
                for (int i = 0; i < 8; ++i) { a[i] = lo[i]; a[i + 8] = hi[i]; }
                const v16h b0 = *(const v16h*)(bp0 + kt * 32);
                const v16h b1 = *(const v16h*)(bp1 + kt * 32);
                acc0a = __builtin_amdgcn_wmma_f32_16x16x32_f16(false, a, false, b0, (short)0, acc0a, false, false);
                acc1a = __builtin_amdgcn_wmma_f32_16x16x32_f16(false, a, false, b1, (short)0, acc1a, false, false);
            }
            {   // odd K-step -> chains b
                const v8h lo = *(const v8h*)(up + (kt + 1) * 32);
                const v8h hi = *(const v8h*)(up + (kt + 1) * 32 + 8);
                v16h a;
#pragma unroll
                for (int i = 0; i < 8; ++i) { a[i] = lo[i]; a[i + 8] = hi[i]; }
                const v16h b0 = *(const v16h*)(bp0 + (kt + 1) * 32);
                const v16h b1 = *(const v16h*)(bp1 + (kt + 1) * 32);
                acc0b = __builtin_amdgcn_wmma_f32_16x16x32_f16(false, a, false, b0, (short)0, acc0b, false, false);
                acc1b = __builtin_amdgcn_wmma_f32_16x16x32_f16(false, a, false, b1, (short)0, acc1b, false, false);
            }
        }
        const v8f acc0 = acc0a + acc0b;
        const v8f acc1 = acc1a + acc1b;

        // ---- epilogue: alpha_{t+1}[m,n] = m_sh[m] + log(V) + logEt[x[b,t+1], n]  (coalesced)
        const int half = lane >> 4;
#pragma unroll
        for (int g = 0; g < 8; ++g) {
            const int mrow = g + half * 8;         // C/D layout: M = g + 8*(lane>=16), N = lane&15
            const float base = m_sh[mrow];
            const float* er  = logEt + (size_t)xt_sh[mrow] * HMM_S;
            alpha_sh[mrow * ASTRIDE + ncol0] = base + __logf(acc0[g]) + er[ncol0];
            alpha_sh[mrow * ASTRIDE + ncol1] = base + __logf(acc1[g]) + er[ncol1];
        }
        __syncthreads();
    }
}

extern "C" void kernel_launch(void* const* d_in, const int* in_sizes, int n_in,
                              void* d_out, int out_size, void* d_ws, size_t ws_size,
                              hipStream_t stream) {
    const int*   x     = (const int*)d_in[0];     // [B, TMAX] int32
    const int*   Tlen  = (const int*)d_in[1];     // [B] int32
    const float* pi    = (const float*)d_in[2];   // [S] f32
    const float* trans = (const float*)d_in[3];   // [S, S] f32
    const float* em    = (const float*)d_in[4];   // [S, O] f32
    float*       out   = (float*)d_out;           // [B, 1] f32

    char* ws = (char*)d_ws;
    float*    logpi = (float*)ws;                                        // 2 KB
    _Float16* Af16  = (_Float16*)(ws + 2048);                            // 512 KB
    float*    logEt = (float*)(ws + 2048 + (size_t)HMM_S * HMM_S * 2);   // 2 MB (transposed [O][S])

    prep_logpi<<<1, 512, 0, stream>>>(pi, logpi);
    prep_trans<<<HMM_S / 16, 512, 0, stream>>>(trans, Af16);
    prep_emis<<<HMM_S, 256, 0, stream>>>(em, logEt);
    hmm_forward<<<HMM_B / 16, 512, 0, stream>>>(x, Tlen, logpi, Af16, logEt, out);
}
